// GRU_84825604096414
// MI455X (gfx1250) — compile-verified, matrix operand order checked by reading, not run on hardware
//
#include <hip/hip_runtime.h>

#define B_   64
#define T_   512
#define I_   256
#define H_   1024
#define O_   23
#define G3H  3072
#define TCH  64                  // timesteps per x_proj chunk
#define NCHUNK (T_ / TCH)

typedef __attribute__((ext_vector_type(16))) __bf16       v16bf;
typedef __attribute__((ext_vector_type(8)))  float        v8f;
typedef __attribute__((ext_vector_type(4)))  unsigned int u32x4;

union Frag { v16bf v; u32x4 q[2]; };

// Load one 16x32 bf16 A/B fragment slice for this lane.
// Per ISA 7.12.2: lane L holds row (L&15); K-base = 8*(L>>4); elements are
// two contiguous 8-element (16B) chunks at +0 and +16 elements.
__device__ __forceinline__ v16bf load_frag(const unsigned short* p) {
  Frag f;
  f.q[0] = *(const u32x4*)(p);
  f.q[1] = *(const u32x4*)(p + 16);
  return f.v;
}

__device__ __forceinline__ unsigned short f2bf(float x) {
  unsigned int u = __float_as_uint(x);
  return (unsigned short)((u + 0x7FFFu + ((u >> 16) & 1u)) >> 16);  // RNE
}

__device__ __forceinline__ float sigmoidf_(float x) {
  return 1.0f / (1.0f + __expf(-x));
}

// ---------------- fp32 -> bf16 conversion ----------------
__global__ void cvt_bf16_kernel(const float* __restrict__ in,
                                unsigned short* __restrict__ out, int n) {
  int i = blockIdx.x * blockDim.x + threadIdx.x;
  int stride = gridDim.x * blockDim.x;
  for (; i < n; i += stride) out[i] = f2bf(in[i]);
}

// ---------------- zero h state (fp32 + bf16) ----------------
__global__ void init_h_kernel(float* __restrict__ hf,
                              unsigned short* __restrict__ hb, int n) {
  int i = blockIdx.x * blockDim.x + threadIdx.x;
  int stride = gridDim.x * blockDim.x;
  for (; i < n; i += stride) { hf[i] = 0.0f; hb[i] = 0; }
}

// ---------------- WMMA GEMM over a time-chunk ----------------
// Wave tile: 32(M) x NT*16(N). Local rows m in [0, B_*TCH):
//   global row grow = (m>>6)*T_ + t0 + (m&63)
// Y[row][n] = sum_k A[grow*K + k] * W[n*K + k] + bias[n]
// OUTG=false: Y row index = m (compact chunk). OUTG=true: Y row = grow.
// Steady state: 2-chunk unroll with ping-pong A-fragment sets (no register
// rotation copies, no WMMA-source WAR hazards). K must be a multiple of 64.
template <int NT, bool OUTG>
__global__ __launch_bounds__(256, 1)
void gemm_kernel(const unsigned short* __restrict__ A,
                 const unsigned short* __restrict__ W,
                 const float* __restrict__ bias,
                 float* __restrict__ Y,
                 int K, int N, int t0) {
  const int wave  = threadIdx.x >> 5;
  const int lane  = threadIdx.x & 31;
  const int l15   = lane & 15;
  const int khalf = (lane >> 4) << 3;          // 0 or 8

  const int m_tile = blockIdx.x * 8 + wave;    // 8 waves/block, 32 rows each
  const int m0     = m_tile * 32;              // 32 | 64 -> tile within one batch b
  const int grow0  = (m0 >> 6) * T_ + t0 + (m0 & 63);
  const int n_blk  = blockIdx.y * (NT * 16);

  const unsigned short* pa0 = A + (size_t)(grow0 + l15) * K + khalf;
  const unsigned short* pa1 = pa0 + (size_t)16 * K;
  const unsigned short* pb[NT];
#pragma unroll
  for (int j = 0; j < NT; ++j) {
    int n  = n_blk + j * 16 + l15;
    int nc = (n < N) ? n : (N - 1);            // clamp: safe load, store guarded
    pb[j] = W + (size_t)nc * K + khalf;
  }

  v8f acc[2][NT];
#pragma unroll
  for (int mt = 0; mt < 2; ++mt)
#pragma unroll
    for (int j = 0; j < NT; ++j) acc[mt][j] = (v8f){0,0,0,0,0,0,0,0};

  // body: consume A frags (ua0,ua1) for chunk kc, optionally prefetch chunk
  // kc+32 into (na0,na1), and load+consume B frags for kc.
  auto body = [&](const v16bf& ua0, const v16bf& ua1,
                  v16bf& na0, v16bf& na1, int kc, bool pref) {
    v16bf b[NT];
#pragma unroll
    for (int j = 0; j < NT; ++j) b[j] = load_frag(pb[j] + kc);
    if (pref) {
      na0 = load_frag(pa0 + kc + 32);
      na1 = load_frag(pa1 + kc + 32);
    }
#pragma unroll
    for (int j = 0; j < NT; ++j) {
      acc[0][j] = __builtin_amdgcn_wmma_f32_16x16x32_bf16(
          false, ua0, false, b[j], (short)0, acc[0][j], false, false);
      acc[1][j] = __builtin_amdgcn_wmma_f32_16x16x32_bf16(
          false, ua1, false, b[j], (short)0, acc[1][j], false, false);
    }
  };

  v16bf s0a0 = load_frag(pa0);                 // set0 <- chunk 0
  v16bf s0a1 = load_frag(pa1);
  v16bf s1a0, s1a1;

#pragma unroll 1
  for (int k0 = 0; k0 < K - 64; k0 += 64) {
    body(s0a0, s0a1, s1a0, s1a1, k0,      true);   // use set0, prefetch set1
    body(s1a0, s1a1, s0a0, s0a1, k0 + 32, true);   // use set1, prefetch set0
  }
  body(s0a0, s0a1, s1a0, s1a1, K - 64, true);      // epilogue pair
  body(s1a0, s1a1, s0a0, s0a1, K - 32, false);

  // --- store (cold path, guarded) ---
#pragma unroll
  for (int j = 0; j < NT; ++j) {
    int n = n_blk + j * 16 + l15;
    if (n >= N) continue;
    float bv = bias ? bias[n] : 0.0f;
#pragma unroll
    for (int mt = 0; mt < 2; ++mt) {
#pragma unroll
      for (int r = 0; r < 8; ++r) {
        int mrow = mt * 16 + r + ((lane >> 4) << 3);  // C/D: M = r + 8*(lane>=16)
        size_t row_out = OUTG ? (size_t)(grow0 + mrow) : (size_t)(m0 + mrow);
        Y[row_out * (size_t)N + n] = acc[mt][j][r] + bv;
      }
    }
  }
}

// ---------------- one fused GRU timestep ----------------
// grid = H_/64 workgroups; each WG owns 64 hidden columns across all 3 gates.
// gh = h_in @ W_hh^T via WMMA, then gate math + h update, ping-pong h buffers.
__global__ __launch_bounds__(256, 1)
void gru_step_kernel(const float* __restrict__ xproj,          // [B_*TCH, 3H] local rows
                     const unsigned short* __restrict__ Whh,   // [3H, H] bf16
                     const float* __restrict__ bhh,            // [3H]
                     const unsigned short* __restrict__ h_in_bf,
                     const float* __restrict__ h_in_f,
                     unsigned short* __restrict__ h_out_bf,
                     float* __restrict__ h_out_f,
                     unsigned short* __restrict__ out_seq,     // [B_*T_, H] bf16
                     int t, int lt) {
  const int wave  = threadIdx.x >> 5;
  const int lane  = threadIdx.x & 31;
  const int l15   = lane & 15;
  const int khalf = (lane >> 4) << 3;
  const int mhalf = wave & 1;                            // batch rows 0-31 / 32-63
  const int c0    = blockIdx.x * 64 + (wave >> 1) * 16;  // hidden column tile

  const unsigned short* pa0 = h_in_bf + (size_t)(mhalf * 32 + l15) * H_ + khalf;
  const unsigned short* pa1 = pa0 + (size_t)16 * H_;
  const unsigned short* pb[3];
#pragma unroll
  for (int g = 0; g < 3; ++g)
    pb[g] = Whh + (size_t)(g * H_ + c0 + l15) * H_ + khalf;

  v8f acc[2][3];
#pragma unroll
  for (int mt = 0; mt < 2; ++mt)
#pragma unroll
    for (int g = 0; g < 3; ++g) acc[mt][g] = (v8f){0,0,0,0,0,0,0,0};

  auto body = [&](const v16bf& ua0, const v16bf& ua1,
                  v16bf& na0, v16bf& na1, int kc, bool pref) {
    v16bf b[3];
#pragma unroll
    for (int g = 0; g < 3; ++g) b[g] = load_frag(pb[g] + kc);
    if (pref) {
      na0 = load_frag(pa0 + kc + 32);
      na1 = load_frag(pa1 + kc + 32);
    }
#pragma unroll
    for (int g = 0; g < 3; ++g) {
      acc[0][g] = __builtin_amdgcn_wmma_f32_16x16x32_bf16(
          false, ua0, false, b[g], (short)0, acc[0][g], false, false);
      acc[1][g] = __builtin_amdgcn_wmma_f32_16x16x32_bf16(
          false, ua1, false, b[g], (short)0, acc[1][g], false, false);
    }
  };

  v16bf s0a0 = load_frag(pa0);                 // set0 <- chunk 0
  v16bf s0a1 = load_frag(pa1);
  v16bf s1a0, s1a1;

#pragma unroll 1
  for (int k0 = 0; k0 < H_ - 64; k0 += 64) {
    body(s0a0, s0a1, s1a0, s1a1, k0,      true);
    body(s1a0, s1a1, s0a0, s0a1, k0 + 32, true);
  }
  body(s0a0, s0a1, s1a0, s1a1, H_ - 64, true);
  body(s1a0, s1a1, s0a0, s0a1, H_ - 32, false);

  // --- gate math + h update ---
  const int j  = c0 + l15;
  const float br = bhh[j];
  const float bz = bhh[H_ + j];
  const float bn = bhh[2 * H_ + j];

#pragma unroll
  for (int mt = 0; mt < 2; ++mt) {
#pragma unroll
    for (int r = 0; r < 8; ++r) {
      int b = mhalf * 32 + mt * 16 + r + ((lane >> 4) << 3);   // batch index
      const float* xp = xproj + (size_t)(b * TCH + lt) * G3H;
      float xr = xp[j];
      float xz = xp[H_ + j];
      float xn = xp[2 * H_ + j];
      float rg = sigmoidf_(xr + acc[mt][0][r] + br);
      float zg = sigmoidf_(xz + acc[mt][1][r] + bz);
      float ng = tanhf(xn + rg * (acc[mt][2][r] + bn));
      float hp = h_in_f[(size_t)b * H_ + j];
      float hn = (1.0f - zg) * ng + zg * hp;
      h_out_f[(size_t)b * H_ + j] = hn;
      unsigned short hbv = f2bf(hn);
      h_out_bf[(size_t)b * H_ + j] = hbv;
      out_seq[((size_t)b * T_ + t) * H_ + j] = hbv;
    }
  }
}

extern "C" void kernel_launch(void* const* d_in, const int* in_sizes, int n_in,
                              void* d_out, int out_size, void* d_ws, size_t ws_size,
                              hipStream_t stream) {
  const float* x    = (const float*)d_in[0];
  const float* Wih0 = (const float*)d_in[1];
  const float* Whh0 = (const float*)d_in[2];
  const float* bih0 = (const float*)d_in[3];
  const float* bhh0 = (const float*)d_in[4];
  const float* Wih1 = (const float*)d_in[5];
  const float* Whh1 = (const float*)d_in[6];
  const float* bih1 = (const float*)d_in[7];
  const float* bhh1 = (const float*)d_in[8];
  const float* fcw  = (const float*)d_in[9];
  const float* fcb  = (const float*)d_in[10];
  float* out = (float*)d_out;

  // ---- carve workspace (≈220 MB) ----
  char* ws = (char*)d_ws;
  size_t off = 0;
  auto carve = [&](size_t bytes) -> void* {
    off = (off + 255) & ~(size_t)255;
    void* p = ws + off;
    off += bytes;
    return p;
  };
  unsigned short* xbf   = (unsigned short*)carve((size_t)B_ * T_ * I_ * 2);
  unsigned short* wih0b = (unsigned short*)carve((size_t)G3H * I_ * 2);
  unsigned short* whh0b = (unsigned short*)carve((size_t)G3H * H_ * 2);
  unsigned short* wih1b = (unsigned short*)carve((size_t)G3H * H_ * 2);
  unsigned short* whh1b = (unsigned short*)carve((size_t)G3H * H_ * 2);
  unsigned short* fcwb  = (unsigned short*)carve((size_t)O_ * H_ * 2);
  unsigned short* out0b = (unsigned short*)carve((size_t)B_ * T_ * H_ * 2);
  unsigned short* out1b = (unsigned short*)carve((size_t)B_ * T_ * H_ * 2);
  float*          xproj = (float*)carve((size_t)B_ * TCH * G3H * 4);
  float*          hf0   = (float*)carve((size_t)B_ * H_ * 4);
  float*          hf1   = (float*)carve((size_t)B_ * H_ * 4);
  unsigned short* hb0   = (unsigned short*)carve((size_t)B_ * H_ * 2);
  unsigned short* hb1   = (unsigned short*)carve((size_t)B_ * H_ * 2);
  float*          hf[2] = {hf0, hf1};
  unsigned short* hb[2] = {hb0, hb1};

  // ---- convert inputs/weights to bf16 ----
  cvt_bf16_kernel<<<1024, 256, 0, stream>>>(x,    xbf,   B_ * T_ * I_);
  cvt_bf16_kernel<<<512,  256, 0, stream>>>(Wih0, wih0b, G3H * I_);
  cvt_bf16_kernel<<<1024, 256, 0, stream>>>(Whh0, whh0b, G3H * H_);
  cvt_bf16_kernel<<<1024, 256, 0, stream>>>(Wih1, wih1b, G3H * H_);
  cvt_bf16_kernel<<<1024, 256, 0, stream>>>(Whh1, whh1b, G3H * H_);
  cvt_bf16_kernel<<<64,   256, 0, stream>>>(fcw,  fcwb,  O_ * H_);

  // ---- two GRU layers ----
  for (int layer = 0; layer < 2; ++layer) {
    const unsigned short* inbf = layer ? out0b : xbf;
    const int K                = layer ? H_ : I_;
    const unsigned short* wih  = layer ? wih1b : wih0b;
    const unsigned short* whh  = layer ? whh1b : whh0b;
    const float* bih           = layer ? bih1 : bih0;
    const float* bhh           = layer ? bhh1 : bhh0;
    unsigned short* oseq       = layer ? out1b : out0b;

    init_h_kernel<<<64, 256, 0, stream>>>(hf[0], hb[0], B_ * H_);

    for (int c = 0; c < NCHUNK; ++c) {
      const int t0 = c * TCH;
      // x-projection for this time chunk: [B_*TCH, 3H]
      gemm_kernel<4, false><<<dim3(16, 48), 256, 0, stream>>>(
          inbf, wih, bih, xproj, K, G3H, t0);
      // sequential scan over the chunk
      for (int lt = 0; lt < TCH; ++lt) {
        const int t = t0 + lt;
        gru_step_kernel<<<H_ / 64, 256, 0, stream>>>(
            xproj, whh, bhh,
            hb[t & 1], hf[t & 1], hb[(t + 1) & 1], hf[(t + 1) & 1],
            oseq, t, lt);
      }
    }
  }

  // ---- final FC: [B*T, H] @ fc_w^T + fc_b -> d_out [B*T, 23] ----
  for (int c = 0; c < NCHUNK; ++c) {
    gemm_kernel<2, true><<<dim3(16, 1), 256, 0, stream>>>(
        out1b, fcwb, fcb, out, H_, O_, c * TCH);
  }
}
